// LlamaAttention_20418274525397
// MI455X (gfx1250) — compile-verified, matrix operand order checked by reading, not run on hardware
//
#include <hip/hip_runtime.h>

// ---------------------------------------------------------------- types
typedef __attribute__((ext_vector_type(8)))  float  v8f;
typedef __attribute__((ext_vector_type(8)))  __bf16 v8bf;
typedef __attribute__((ext_vector_type(16))) __bf16 v16bf;

// ---------------------------------------------------------------- sizes
constexpr int S   = 2048;
constexpr int H   = 4096;
constexpr int NH  = 32;
constexpr int NKV = 8;
constexpr int HD  = 128;
constexpr int NQ  = NH * HD;    // 4096
constexpr int NKD = NKV * HD;   // 1024
constexpr float QSCALE = 0.08838834764831845f;   // 1/sqrt(128)
constexpr float LN_ROPE = 9.210340371976184f;    // ln(10000)

// workspace byte offsets
constexpr size_t OFF_XBF = 0;
constexpr size_t SZ_XBF  = (size_t)S * H * 2;
constexpr size_t OFF_WQT = OFF_XBF + SZ_XBF;
constexpr size_t SZ_WQT  = (size_t)H * NQ * 2;
constexpr size_t OFF_WKT = OFF_WQT + SZ_WQT;
constexpr size_t SZ_WKT  = (size_t)H * NKD * 2;
constexpr size_t OFF_WVT = OFF_WKT + SZ_WKT;
constexpr size_t SZ_WVT  = SZ_WKT;
constexpr size_t OFF_WOT = OFF_WVT + SZ_WVT;
constexpr size_t SZ_WOT  = (size_t)NQ * H * 2;
constexpr size_t OFF_QR  = OFF_WOT + SZ_WOT;
constexpr size_t SZ_QR   = (size_t)NH * S * HD * 2;
constexpr size_t OFF_KR  = OFF_QR + SZ_QR;
constexpr size_t SZ_KR   = (size_t)NKV * S * HD * 2;
constexpr size_t OFF_VT  = OFF_KR + SZ_KR;
constexpr size_t SZ_VT   = SZ_KR;
constexpr size_t OFF_ATT = OFF_VT + SZ_VT;
constexpr size_t SZ_ATT  = (size_t)S * NQ * 2;   // total ~143 MB

// ---------------------------------------------------------------- helpers
static __device__ __forceinline__ v8f wmma_bf16(v16bf a, v16bf b, v8f c) {
    return __builtin_amdgcn_wmma_f32_16x16x32_bf16(
        /*neg_a=*/false, a, /*neg_b=*/false, b,
        /*c_mod=*/(short)0, c, /*reuse_a=*/false, /*reuse_b=*/false);
}

// Build a 16x32 bf16 fragment (A layout; B layout identical with N for M).
// `p` points at this lane's row start within the K-window; lanes 0-15 take
// K {0..7,16..23}, lanes 16-31 take K {8..15,24..31}.
static __device__ __forceinline__ v16bf ldfrag(const __bf16* p, int g) {
    v8bf lo = *reinterpret_cast<const v8bf*>(p + 8 * g);
    v8bf hi = *reinterpret_cast<const v8bf*>(p + 16 + 8 * g);
    v16bf r;
#pragma unroll
    for (int i = 0; i < 8; ++i) { r[i] = lo[i]; r[i + 8] = hi[i]; }
    return r;
}

// ---------------------------------------------------------------- prep
__global__ void cvt_bf16_kernel(const float* __restrict__ in,
                                __bf16* __restrict__ out, size_t n) {
    size_t stride = (size_t)gridDim.x * blockDim.x;
    for (size_t i = (size_t)blockIdx.x * blockDim.x + threadIdx.x; i < n; i += stride)
        out[i] = (__bf16)in[i];
}

// in: [K][N] f32, out: [N][K] bf16 (writes coalesced along K)
__global__ void cvt_transpose_kernel(const float* __restrict__ in,
                                     __bf16* __restrict__ out, int K, int N) {
    size_t total = (size_t)K * N;
    size_t stride = (size_t)gridDim.x * blockDim.x;
    for (size_t i = (size_t)blockIdx.x * blockDim.x + threadIdx.x; i < total; i += stride) {
        size_t n = i / (size_t)K, k = i % (size_t)K;
        out[i] = (__bf16)in[k * (size_t)N + n];
    }
}

// ---------------------------------------------------------------- QKV + RoPE
// grid: (S/32, NH + 2*NKV), block: 32.  Each wave computes a 32x128 tile
// (one head-worth of columns, two 16-row blocks sharing every B fragment),
// applies RoPE in-register, stores bf16.
__global__ void qkv_rope_kernel(const __bf16* __restrict__ Xbf,
                                const __bf16* __restrict__ Wqt,
                                const __bf16* __restrict__ Wkt,
                                const __bf16* __restrict__ Wvt,
                                __bf16* __restrict__ Qr,
                                __bf16* __restrict__ Kr,
                                __bf16* __restrict__ Vt) {
    const int mb   = blockIdx.x * 32;       // row base
    const int slot = blockIdx.y;
    const int lane = threadIdx.x;
    const int g    = lane >> 4;
    const int q15  = lane & 15;

    int mode, head;
    const __bf16* Wt;
    if (slot < NH)            { mode = 0; head = slot;            Wt = Wqt; }
    else if (slot < NH + NKV) { mode = 1; head = slot - NH;       Wt = Wkt; }
    else                      { mode = 2; head = slot - NH - NKV; Wt = Wvt; }

    const __bf16* arow0 = Xbf + (size_t)(mb + q15) * H;
    const __bf16* arow1 = arow0 + (size_t)16 * H;
    const __bf16* brow  = Wt + (size_t)(head * HD + q15) * H;

    // acc[t]   : rows mb..mb+15,  cols head*HD + t*16 ..
    // acc[t+8] : rows mb+16..mb+31
    v8f acc[16];
#pragma unroll
    for (int t = 0; t < 16; ++t) acc[t] = (v8f)0.0f;

    for (int k0 = 0; k0 < H; k0 += 32) {
        v16bf a0 = ldfrag(arow0 + k0, g);
        v16bf a1 = ldfrag(arow1 + k0, g);
#pragma unroll
        for (int t = 0; t < 8; ++t) {
            v16bf bf = ldfrag(brow + (size_t)t * 16 * H + k0, g);
            acc[t]     = wmma_bf16(a0, bf, acc[t]);
            acc[t + 8] = wmma_bf16(a1, bf, acc[t + 8]);
        }
    }

    if (mode == 2) {                      // V: store transposed [h][d][s]
        __bf16* dst = Vt + (size_t)head * HD * S;
#pragma unroll
        for (int rb = 0; rb < 2; ++rb)
#pragma unroll
            for (int t = 0; t < 8; ++t)
#pragma unroll
                for (int r = 0; r < 8; ++r)
                    dst[(size_t)(t * 16 + q15) * S + (mb + rb * 16 + r + 8 * g)] =
                        (__bf16)acc[t + 8 * rb][r];
        return;
    }

    // RoPE: column d = t*16+q15; pair (d, d+64) == tiles (t, t+4), same lane.
    const float scale = (mode == 0) ? QSCALE : 1.0f;
#pragma unroll
    for (int tp = 0; tp < 4; ++tp) {
        const int j = tp * 16 + q15;                 // d % 64
        const float invf = __expf(-(float)j * (LN_ROPE / 64.0f));
#pragma unroll
        for (int rb = 0; rb < 2; ++rb)
#pragma unroll
            for (int r = 0; r < 8; ++r) {
                const float pos = (float)(mb + rb * 16 + r + 8 * g);
                float sn, cs;
                __sincosf(pos * invf, &sn, &cs);
                const float x1 = acc[tp + 8 * rb][r], x2 = acc[tp + 4 + 8 * rb][r];
                acc[tp + 8 * rb][r]     = (x1 * cs - x2 * sn) * scale;
                acc[tp + 4 + 8 * rb][r] = (x2 * cs + x1 * sn) * scale;
            }
    }

    __bf16* dst = (mode == 0 ? Qr : Kr) + (size_t)head * S * HD;
#pragma unroll
    for (int rb = 0; rb < 2; ++rb)
#pragma unroll
        for (int t = 0; t < 8; ++t)
#pragma unroll
            for (int r = 0; r < 8; ++r)
                dst[(size_t)(mb + rb * 16 + r + 8 * g) * HD + t * 16 + q15] =
                    (__bf16)acc[t + 8 * rb][r];
}

// ---------------------------------------------------------------- attention
// grid: (S/64, NH), block: 128 = 4 waves.  The 4 waves handle 4 consecutive
// 16-row q-tiles of the SAME head, so they stream identical K/V cachelines
// through the WGP's L0 cache.  No barriers: each wave owns a private LDS
// slice for its P tile (same-wave DS ordering covers the store->load dep).
__global__ void attn_kernel(const __bf16* __restrict__ Qr,
                            const __bf16* __restrict__ Kr,
                            const __bf16* __restrict__ Vt,
                            __bf16* __restrict__ Attn) {
    __shared__ __bf16 PldsAll[4][16 * 32];

    const int wid  = threadIdx.x >> 5;
    const int lane = threadIdx.x & 31;
    const int mt   = blockIdx.x * 4 + wid;
    const int h    = blockIdx.y;
    const int kvh  = h >> 2;               // NH/NKV = 4
    const int g    = lane >> 4;
    const int q15  = lane & 15;
    const int qb   = mt * 16;
    __bf16* Plds   = PldsAll[wid];

    // Q fragments for this wave's 16 rows (K = HD = 128 -> 4 fragments)
    const __bf16* qrow = Qr + (size_t)h * S * HD + (size_t)(qb + q15) * HD;
    v16bf qf[4];
#pragma unroll
    for (int f = 0; f < 4; ++f) qf[f] = ldfrag(qrow + f * 32, g);

    v8f acc[8];
    float mrow[8], lrow[8];
#pragma unroll
    for (int t = 0; t < 8; ++t) acc[t] = (v8f)0.0f;
#pragma unroll
    for (int r = 0; r < 8; ++r) { mrow[r] = -3.0e38f; lrow[r] = 0.0f; }

    const __bf16* Kh = Kr + (size_t)kvh * S * HD;
    const __bf16* Vh = Vt + (size_t)kvh * HD * S;
    const int qEnd = qb + 15;

    for (int kb = 0; kb <= qEnd; kb += 32) {
        // scores: two 16x16 tiles (keys kb..kb+15, kb+16..kb+31)
        v8f s0 = (v8f)0.0f, s1 = (v8f)0.0f;
        const __bf16* k0row = Kh + (size_t)(kb + q15) * HD;
        const __bf16* k1row = Kh + (size_t)(kb + 16 + q15) * HD;
#pragma unroll
        for (int f = 0; f < 4; ++f) {
            s0 = wmma_bf16(qf[f], ldfrag(k0row + f * 32, g), s0);
            s1 = wmma_bf16(qf[f], ldfrag(k1row + f * 32, g), s1);
        }

        const int key0 = kb + q15, key1 = kb + 16 + q15;
#pragma unroll
        for (int r = 0; r < 8; ++r) {
            const int srow = qb + r + 8 * g;
            float a = (key0 <= srow) ? s0[r] : -1.0e9f;
            float b = (key1 <= srow) ? s1[r] : -1.0e9f;

            // row max over 32 keys (16 lanes of this half x 2 tiles)
            float mx = fmaxf(a, b);
#pragma unroll
            for (int off = 8; off >= 1; off >>= 1)
                mx = fmaxf(mx, __shfl_xor(mx, off, 32));
            const float mnew = fmaxf(mrow[r], mx);
            const float corr = __expf(mrow[r] - mnew);
            const float p0 = __expf(a - mnew);
            const float p1 = __expf(b - mnew);
            float sum = p0 + p1;
#pragma unroll
            for (int off = 8; off >= 1; off >>= 1)
                sum += __shfl_xor(sum, off, 32);
            lrow[r] = lrow[r] * corr + sum;
            mrow[r] = mnew;
#pragma unroll
            for (int t = 0; t < 8; ++t) acc[t][r] *= corr;

            Plds[(r + 8 * g) * 32 + q15]      = (__bf16)p0;
            Plds[(r + 8 * g) * 32 + 16 + q15] = (__bf16)p1;
        }

        // P (16x32) as A-fragment, V-block as B-fragments
        v16bf pf = ldfrag(Plds + q15 * 32, g);
#pragma unroll
        for (int t = 0; t < 8; ++t) {
            v16bf vb = ldfrag(Vh + (size_t)(t * 16 + q15) * S + kb, g);
            acc[t] = wmma_bf16(pf, vb, acc[t]);
        }
    }

    // normalize and store bf16 into [s][h*HD + d]
#pragma unroll
    for (int r = 0; r < 8; ++r) {
        const float inv = 1.0f / lrow[r];
        const size_t row = (size_t)(qb + r + 8 * g) * NQ + h * HD;
#pragma unroll
        for (int t = 0; t < 8; ++t)
            Attn[row + t * 16 + q15] = (__bf16)(acc[t][r] * inv);
    }
}

// ---------------------------------------------------------------- out proj
// grid: (S/32, NQ/64), block: 32.  32x64 f32 tile per wave; every B
// fragment is reused by both 16-row blocks.
__global__ void out_gemm_kernel(const __bf16* __restrict__ Attn,
                                const __bf16* __restrict__ Wot,
                                float* __restrict__ Out) {
    const int mb   = blockIdx.x * 32;
    const int nt   = blockIdx.y;
    const int lane = threadIdx.x;
    const int g    = lane >> 4;
    const int q15  = lane & 15;

    const __bf16* arow0 = Attn + (size_t)(mb + q15) * NQ;
    const __bf16* arow1 = arow0 + (size_t)16 * NQ;
    const __bf16* brow  = Wot + (size_t)(nt * 64 + q15) * NQ;

    v8f acc[8];
#pragma unroll
    for (int t = 0; t < 8; ++t) acc[t] = (v8f)0.0f;

    for (int k0 = 0; k0 < NQ; k0 += 32) {
        v16bf a0 = ldfrag(arow0 + k0, g);
        v16bf a1 = ldfrag(arow1 + k0, g);
#pragma unroll
        for (int t = 0; t < 4; ++t) {
            v16bf bf = ldfrag(brow + (size_t)t * 16 * NQ + k0, g);
            acc[t]     = wmma_bf16(a0, bf, acc[t]);
            acc[t + 4] = wmma_bf16(a1, bf, acc[t + 4]);
        }
    }

#pragma unroll
    for (int rb = 0; rb < 2; ++rb)
#pragma unroll
        for (int r = 0; r < 8; ++r) {
            const size_t row = (size_t)(mb + rb * 16 + r + 8 * g) * H + nt * 64;
#pragma unroll
            for (int t = 0; t < 4; ++t)
                Out[row + t * 16 + q15] = acc[t + 4 * rb][r];
        }
}

// ---------------------------------------------------------------- launch
extern "C" void kernel_launch(void* const* d_in, const int* in_sizes, int n_in,
                              void* d_out, int out_size, void* d_ws, size_t ws_size,
                              hipStream_t stream) {
    const float* X  = (const float*)d_in[0];
    const float* Wq = (const float*)d_in[3];
    const float* Wk = (const float*)d_in[4];
    const float* Wv = (const float*)d_in[5];
    const float* Wo = (const float*)d_in[6];
    float* Out = (float*)d_out;

    char* ws = (char*)d_ws;
    __bf16* Xbf = (__bf16*)(ws + OFF_XBF);
    __bf16* Wqt = (__bf16*)(ws + OFF_WQT);
    __bf16* Wkt = (__bf16*)(ws + OFF_WKT);
    __bf16* Wvt = (__bf16*)(ws + OFF_WVT);
    __bf16* Wot = (__bf16*)(ws + OFF_WOT);
    __bf16* Qr  = (__bf16*)(ws + OFF_QR);
    __bf16* Kr  = (__bf16*)(ws + OFF_KR);
    __bf16* Vt  = (__bf16*)(ws + OFF_VT);
    __bf16* Att = (__bf16*)(ws + OFF_ATT);

    auto blocks = [](size_t n) { return (unsigned)((n + 255) / 256); };

    cvt_bf16_kernel<<<blocks((size_t)S * H), 256, 0, stream>>>(X, Xbf, (size_t)S * H);
    cvt_transpose_kernel<<<blocks((size_t)H * NQ),  256, 0, stream>>>(Wq, Wqt, H, NQ);
    cvt_transpose_kernel<<<blocks((size_t)H * NKD), 256, 0, stream>>>(Wk, Wkt, H, NKD);
    cvt_transpose_kernel<<<blocks((size_t)H * NKD), 256, 0, stream>>>(Wv, Wvt, H, NKD);
    cvt_transpose_kernel<<<blocks((size_t)NQ * H),  256, 0, stream>>>(Wo, Wot, NQ, H);

    qkv_rope_kernel<<<dim3(S / 32, NH + 2 * NKV), 32, 0, stream>>>(
        Xbf, Wqt, Wkt, Wvt, Qr, Kr, Vt);

    attn_kernel<<<dim3(S / 64, NH), 128, 0, stream>>>(Qr, Kr, Vt, Att);

    out_gemm_kernel<<<dim3(S / 32, NQ / 64), 32, 0, stream>>>(Att, Wot, Out);
}
